// Interaction_1829656068124
// MI455X (gfx1250) — matrix operand order repaired and checked by reading, not compiled, not run on hardware
//
#include <hip/hip_runtime.h>
#include <hip/hip_bf16.h>
#include <math.h>

// ---------------------------------------------------------------------------
// SchNet interaction block on gfx1250 (MI455X), wave32 + WMMA bf16.
// ---------------------------------------------------------------------------

#define NNODES 50000
#define NEDGES 1000000
#define NF     128
#define NRBF   51
#define GAMMA  10.0f
#define RBFSTEP 0.1f
#define LOG2F_ 0.6931471805599453f
#define WPB    8           // waves per block (256 threads)

#define WF1_ELEMS (8 * 2 * 32 * 16)   // 8192  (K padded 51->64)
#define WF2_ELEMS (8 * 4 * 32 * 16)   // 16384

typedef __attribute__((ext_vector_type(16))) __bf16 v16bf;
typedef __attribute__((ext_vector_type(8)))  __bf16 v8bf;
typedef __attribute__((ext_vector_type(8)))  float  v8f;

__device__ __forceinline__ float ssp_f(float x) {
    // shifted softplus, numerically stable
    return fmaxf(x, 0.0f) + log1pf(__expf(-fabsf(x))) - LOG2F_;
}

// A-fragment K offset for element i (16-bit A 16x32 layout):
// lanes 0-15 hold K {0..7, 16..23}; lanes 16-31 hold K {8..15, 24..31}
__device__ __forceinline__ int a_koff(int lane, int i) {
    return ((lane >> 4) << 3) + (i < 8 ? i : i + 8);
}

// Build a v16bf A-fragment from a 16B-aligned row in LDS/regs via two packed
// 8-element chunks (emits 2x ds_load_b128 instead of 16x ds_load_u16).
__device__ __forceinline__ v16bf load_a_frag_lds(const __bf16* rowBase) {
    v8bf lo = *(const v8bf*)(rowBase);
    v8bf hi = *(const v8bf*)(rowBase + 16);
    v16bf a;
    #pragma unroll
    for (int i = 0; i < 8; ++i) { a[i] = lo[i]; a[8 + i] = hi[i]; }
    return a;
}

// ---------------------------------------------------------------------------
// Weight prep: f32 row-major [Ksrc x 128] -> bf16 B-fragments, K padded to
// nkt*32. Fragment layout: dst[((nt*nkt + kt)*32 + lane)*16 + i] =
//   B[kt*32 + (lane>>4)*8 + (i < 8 ? i : i + 8)][nt*16 + (lane&15)]
// so a wave loads one whole B fragment as one contiguous v16bf per lane.
// ---------------------------------------------------------------------------
__global__ void prep_weights_kernel(const float* __restrict__ src,
                                    __bf16* __restrict__ dst,
                                    int Ksrc, int nkt) {
    int idx = blockIdx.x * blockDim.x + threadIdx.x;
    int total = 8 * nkt * 32 * 16;
    if (idx >= total) return;
    int i    = idx & 15;
    int lane = (idx >> 4) & 31;
    int kt   = (idx >> 9) % nkt;
    int nt   = idx / (nkt << 9);
    int kRow = kt * 32 + ((lane >> 4) << 3) + (i < 8 ? i : i + 8);
    int col  = nt * 16 + (lane & 15);
    float vv = (kRow < Ksrc) ? src[kRow * NF + col] : 0.0f;
    dst[idx] = (__bf16)vv;
}

__global__ void zero_kernel(float* __restrict__ p, int n) {
    int i = blockIdx.x * blockDim.x + threadIdx.x;
    if (i < n) p[i] = 0.0f;
}

// ---------------------------------------------------------------------------
// m = x @ W_lin1 + b_lin1   (one wave = 16 rows x 128 cols)
// ---------------------------------------------------------------------------
__global__ __launch_bounds__(256) void node_lin1_kernel(
    const float* __restrict__ x, const __bf16* __restrict__ wfrag,
    const float* __restrict__ bias, float* __restrict__ m) {
    const int lane = threadIdx.x & 31;
    const int wave = threadIdx.x >> 5;
    const int tile = blockIdx.x * WPB + wave;
    if (tile * 16 >= NNODES) return;
    const int r0  = tile * 16;
    const int row = r0 + (lane & 15);
    const int kb  = (lane >> 4) << 3;

    v16bf a[4];
    #pragma unroll
    for (int kt = 0; kt < 4; ++kt) {
        const float* p = x + (size_t)row * NF + kt * 32 + kb;
        #pragma unroll
        for (int i = 0; i < 8; ++i) a[kt][i]     = (__bf16)p[i];
        #pragma unroll
        for (int i = 0; i < 8; ++i) a[kt][8 + i] = (__bf16)p[16 + i];
    }
    v8f c[8];
    #pragma unroll
    for (int n = 0; n < 8; ++n) c[n] = (v8f){0,0,0,0,0,0,0,0};
    #pragma unroll
    for (int n = 0; n < 8; ++n)
        #pragma unroll
        for (int kt = 0; kt < 4; ++kt) {
            v16bf b = *(const v16bf*)(wfrag + ((((n << 2) + kt) * 32 + lane) << 4));
            c[n] = __builtin_amdgcn_wmma_f32_16x16x32_bf16(
                false, a[kt], false, b, (short)0, c[n], false, false);
        }
    const int colB  = lane & 15;
    const int rBase = (lane >> 4) << 3;
    #pragma unroll
    for (int n = 0; n < 8; ++n) {
        const int f = colB + 16 * n;
        const float bb = bias[f];
        #pragma unroll
        for (int r = 0; r < 8; ++r)
            m[(size_t)(r0 + rBase + r) * NF + f] = c[n][r] + bb;
    }
}

// ---------------------------------------------------------------------------
// Edge kernel: RBF -> ssp(rbf@Wf1+bf1) -> @Wf2+bf2 -> msg = m[src]*W ->
// atomic scatter into v[dst].  One wave = 16 edges.
// Wf1/Wf2 fragments staged in LDS once per block (immune to gather thrash).
// ---------------------------------------------------------------------------
__global__ __launch_bounds__(256) void edge_cfconv_kernel(
    const int* __restrict__ edge_index, const float* __restrict__ pos,
    const float* __restrict__ m, const __bf16* __restrict__ wf1,
    const float* __restrict__ bf1, const __bf16* __restrict__ wf2,
    const float* __restrict__ bf2, float* __restrict__ v) {
    __shared__ __bf16 sWf1[WF1_ELEMS];          // 16 KB
    __shared__ __bf16 sWf2[WF2_ELEMS];          // 32 KB
    __shared__ __bf16 stage[WPB * 16 * NF];     // 32 KB

    // cooperative weight preload (16B per thread per iter)
    {
        const uint4* s1 = (const uint4*)wf1;
        uint4*       d1 = (uint4*)sWf1;
        for (int i = threadIdx.x; i < WF1_ELEMS / 8; i += 256) d1[i] = s1[i];
        const uint4* s2 = (const uint4*)wf2;
        uint4*       d2 = (uint4*)sWf2;
        for (int i = threadIdx.x; i < WF2_ELEMS / 8; i += 256) d2[i] = s2[i];
    }
    __syncthreads();

    const int lane = threadIdx.x & 31;
    const int wave = threadIdx.x >> 5;
    const int tile = blockIdx.x * WPB + wave;
    if (tile * 16 >= NEDGES) return;   // wave-uniform exit (after barrier)
    const int e0 = tile * 16;
    const int er = lane & 15;

    // each half-wave redundantly loads its edge's endpoints + distance
    const int s_idx = edge_index[e0 + er];
    const int d_idx = edge_index[NEDGES + e0 + er];
    const float dx = pos[d_idx * 3 + 0] - pos[s_idx * 3 + 0];
    const float dy = pos[d_idx * 3 + 1] - pos[s_idx * 3 + 1];
    const float dz = pos[d_idx * 3 + 2] - pos[s_idx * 3 + 2];
    const float d  = sqrtf(dx * dx + dy * dy + dz * dz + 1e-12f);

    // Gaussian RBF expansion written straight into A-fragment layout (16x64)
    v16bf a1[2];
    #pragma unroll
    for (int kt = 0; kt < 2; ++kt)
        #pragma unroll
        for (int i = 0; i < 16; ++i) {
            int k = kt * 32 + a_koff(lane, i);
            float t = d - RBFSTEP * (float)k;
            a1[kt][i] = (__bf16)__expf(-GAMMA * t * t);
        }

    // layer 1: [16x64] @ [64x128]
    v8f c1[8];
    #pragma unroll
    for (int n = 0; n < 8; ++n) c1[n] = (v8f){0,0,0,0,0,0,0,0};
    #pragma unroll
    for (int n = 0; n < 8; ++n)
        #pragma unroll
        for (int kt = 0; kt < 2; ++kt) {
            v16bf b = *(const v16bf*)(sWf1 + ((((n << 1) + kt) * 32 + lane) << 4));
            c1[n] = __builtin_amdgcn_wmma_f32_16x16x32_bf16(
                false, a1[kt], false, b, (short)0, c1[n], false, false);
        }

    // bias + shifted softplus, stage through per-wave LDS tile (C->A relayout)
    __bf16* myLds   = stage + wave * (16 * NF);
    const int colB  = lane & 15;
    const int rBase = (lane >> 4) << 3;
    #pragma unroll
    for (int n = 0; n < 8; ++n) {
        const float bb = bf1[colB + 16 * n];
        #pragma unroll
        for (int r = 0; r < 8; ++r)
            stage[wave * (16 * NF) + (rBase + r) * NF + colB + 16 * n] =
                (__bf16)ssp_f(c1[n][r] + bb);
    }
    // re-read as A fragments (same-wave LDS ops are in order); packed b128 loads
    v16bf a2[4];
    const int kb = (lane >> 4) << 3;
    #pragma unroll
    for (int kt = 0; kt < 4; ++kt)
        a2[kt] = load_a_frag_lds(myLds + er * NF + kt * 32 + kb);

    // layer 2: [16x128] @ [128x128]
    v8f c2[8];
    #pragma unroll
    for (int n = 0; n < 8; ++n) c2[n] = (v8f){0,0,0,0,0,0,0,0};
    #pragma unroll
    for (int n = 0; n < 8; ++n)
        #pragma unroll
        for (int kt = 0; kt < 4; ++kt) {
            v16bf b = *(const v16bf*)(sWf2 + ((((n << 2) + kt) * 32 + lane) << 4));
            c2[n] = __builtin_amdgcn_wmma_f32_16x16x32_bf16(
                false, a2[kt], false, b, (short)0, c2[n], false, false);
        }

    // gather m[src] * W, scatter-add into v[dst] (both L2-resident)
    int sv[8], dv[8];
    #pragma unroll
    for (int r = 0; r < 8; ++r) {
        sv[r] = __shfl(s_idx, rBase + r, 32);
        dv[r] = __shfl(d_idx, rBase + r, 32);
    }
    #pragma unroll
    for (int n = 0; n < 8; ++n) {
        const int f = colB + 16 * n;
        const float bb = bf2[f];
        #pragma unroll
        for (int r = 0; r < 8; ++r) {
            const float w   = c2[n][r] + bb;
            const float val = m[(size_t)sv[r] * NF + f] * w;
            atomicAdd(&v[(size_t)dv[r] * NF + f], val);
        }
    }
}

// ---------------------------------------------------------------------------
// out = x + ssp(v@Wm1+bm1)@Wm2 + bm2   (one wave = 16 rows)
// ---------------------------------------------------------------------------
__global__ __launch_bounds__(256) void node_mlp_kernel(
    const float* __restrict__ vin, const float* __restrict__ x,
    const __bf16* __restrict__ wm1, const float* __restrict__ bm1,
    const __bf16* __restrict__ wm2, const float* __restrict__ bm2,
    float* __restrict__ out) {
    __shared__ __bf16 stage[WPB * 16 * NF];
    const int lane = threadIdx.x & 31;
    const int wave = threadIdx.x >> 5;
    const int tile = blockIdx.x * WPB + wave;
    if (tile * 16 >= NNODES) return;
    const int r0  = tile * 16;
    const int row = r0 + (lane & 15);
    const int kb  = (lane >> 4) << 3;

    v16bf a[4];
    #pragma unroll
    for (int kt = 0; kt < 4; ++kt) {
        const float* p = vin + (size_t)row * NF + kt * 32 + kb;
        #pragma unroll
        for (int i = 0; i < 8; ++i) a[kt][i]     = (__bf16)p[i];
        #pragma unroll
        for (int i = 0; i < 8; ++i) a[kt][8 + i] = (__bf16)p[16 + i];
    }
    v8f c1[8];
    #pragma unroll
    for (int n = 0; n < 8; ++n) c1[n] = (v8f){0,0,0,0,0,0,0,0};
    #pragma unroll
    for (int n = 0; n < 8; ++n)
        #pragma unroll
        for (int kt = 0; kt < 4; ++kt) {
            v16bf b = *(const v16bf*)(wm1 + ((((n << 2) + kt) * 32 + lane) << 4));
            c1[n] = __builtin_amdgcn_wmma_f32_16x16x32_bf16(
                false, a[kt], false, b, (short)0, c1[n], false, false);
        }

    __bf16* myLds   = stage + wave * (16 * NF);
    const int colB  = lane & 15;
    const int rBase = (lane >> 4) << 3;
    #pragma unroll
    for (int n = 0; n < 8; ++n) {
        const float bb = bm1[colB + 16 * n];
        #pragma unroll
        for (int r = 0; r < 8; ++r)
            myLds[(rBase + r) * NF + colB + 16 * n] = (__bf16)ssp_f(c1[n][r] + bb);
    }
    v16bf a2[4];
    #pragma unroll
    for (int kt = 0; kt < 4; ++kt)
        a2[kt] = load_a_frag_lds(myLds + (lane & 15) * NF + kt * 32 + kb);

    v8f c2[8];
    #pragma unroll
    for (int n = 0; n < 8; ++n) c2[n] = (v8f){0,0,0,0,0,0,0,0};
    #pragma unroll
    for (int n = 0; n < 8; ++n)
        #pragma unroll
        for (int kt = 0; kt < 4; ++kt) {
            v16bf b = *(const v16bf*)(wm2 + ((((n << 2) + kt) * 32 + lane) << 4));
            c2[n] = __builtin_amdgcn_wmma_f32_16x16x32_bf16(
                false, a2[kt], false, b, (short)0, c2[n], false, false);
        }
    #pragma unroll
    for (int n = 0; n < 8; ++n) {
        const int f = colB + 16 * n;
        const float bb = bm2[f];
        #pragma unroll
        for (int r = 0; r < 8; ++r) {
            const size_t o = (size_t)(r0 + rBase + r) * NF + f;
            out[o] = x[o] + c2[n][r] + bb;
        }
    }
}

// ---------------------------------------------------------------------------
extern "C" void kernel_launch(void* const* d_in, const int* in_sizes, int n_in,
                              void* d_out, int out_size, void* d_ws, size_t ws_size,
                              hipStream_t stream) {
    const float* x          = (const float*)d_in[0];
    const int*   edge_index = (const int*)  d_in[1];
    // d_in[2] = z (unused)
    const float* position   = (const float*)d_in[3];
    const float* W_lin1     = (const float*)d_in[4];
    const float* b_lin1     = (const float*)d_in[5];
    const float* Wf1        = (const float*)d_in[6];
    const float* bf1        = (const float*)d_in[7];
    const float* Wf2        = (const float*)d_in[8];
    const float* bf2        = (const float*)d_in[9];
    const float* Wm1        = (const float*)d_in[10];
    const float* bm1        = (const float*)d_in[11];
    const float* Wm2        = (const float*)d_in[12];
    const float* bm2        = (const float*)d_in[13];
    float* out = (float*)d_out;

    char* ws = (char*)d_ws;
    float*  m     = (float*)ws;                             // 25.6 MB
    float*  v     = (float*)(ws + (size_t)NNODES * NF * 4); // 25.6 MB
    __bf16* wlin1 = (__bf16*)(ws + 2 * (size_t)NNODES * NF * 4);
    __bf16* wf1   = wlin1 + WF2_ELEMS;
    __bf16* wf2   = wf1   + WF1_ELEMS;
    __bf16* wm1   = wf2   + WF2_ELEMS;
    __bf16* wm2   = wm1   + WF2_ELEMS;

    // weight fragment prep (bf16, WMMA B-layout)
    prep_weights_kernel<<<(WF2_ELEMS + 255) / 256, 256, 0, stream>>>(W_lin1, wlin1, 128, 4);
    prep_weights_kernel<<<(WF1_ELEMS + 255) / 256, 256, 0, stream>>>(Wf1,   wf1,  NRBF, 2);
    prep_weights_kernel<<<(WF2_ELEMS + 255) / 256, 256, 0, stream>>>(Wf2,   wf2,  128, 4);
    prep_weights_kernel<<<(WF2_ELEMS + 255) / 256, 256, 0, stream>>>(Wm1,   wm1,  128, 4);
    prep_weights_kernel<<<(WF2_ELEMS + 255) / 256, 256, 0, stream>>>(Wm2,   wm2,  128, 4);

    zero_kernel<<<(NNODES * NF + 255) / 256, 256, 0, stream>>>(v, NNODES * NF);

    const int nodeTiles = NNODES / 16;   // 3125
    const int edgeTiles = NEDGES / 16;   // 62500
    node_lin1_kernel<<<(nodeTiles + WPB - 1) / WPB, 256, 0, stream>>>(x, wlin1, b_lin1, m);
    edge_cfconv_kernel<<<(edgeTiles + WPB - 1) / WPB, 256, 0, stream>>>(
        edge_index, position, m, wf1, bf1, wf2, bf2, v);
    node_mlp_kernel<<<(nodeTiles + WPB - 1) / WPB, 256, 0, stream>>>(
        v, x, wm1, bm1, wm2, bm2, out);
}
